// VisionAttention_36953898615229
// MI455X (gfx1250) — matrix operand order repaired and checked by reading, not compiled, not run on hardware
//
#include <hip/hip_runtime.h>
#include <hip/hip_bf16.h>
#include <math.h>

typedef __bf16 bf16_t;
typedef __attribute__((ext_vector_type(16))) __bf16 v16bf;
typedef __attribute__((ext_vector_type(8)))  float  v8f;

#define NTOK   8192
#define DMODEL 1280
#define NH     16
#define HD     80
#define HDP    96
#define NSEQ   16
#define QSCALE 0.11180339887498948f   // 80^-0.5
#define SPAD   776                    // fp32 score row stride (max L = 768 + pad)
#define PPAD   808                    // bf16 P row stride (conflict-free A-frag reads)

#define WMMA_BF16(a, b, c) \
  __builtin_amdgcn_wmma_f32_16x16x32_bf16(false, (a), false, (b), (short)0, (c), false, false)

__device__ inline v8f vzero8() {
  v8f z;
  #pragma unroll
  for (int i = 0; i < 8; ++i) z[i] = 0.f;
  return z;
}

// CDNA5 async global->LDS copy (ASYNCcnt-tracked); lds_off = workgroup LDS byte offset.
__device__ inline void async_load_b128(unsigned lds_off, const void* gaddr) {
  asm volatile("global_load_async_to_lds_b128 %0, %1, off"
               :: "v"(lds_off), "v"((unsigned long long)gaddr)
               : "memory");
}
__device__ inline void wait_asynccnt0() {
  asm volatile("s_wait_asynccnt 0x0" ::: "memory");
}

// A-matrix fragment (16x32 bf16, row-major source, stride in elements, stride even).
// lane<16 -> M=lane, K=0..15 ; lane>=16 -> M=lane-16, K=16..31
__device__ inline v16bf load_a_frag(const bf16_t* base, int stride, int lane) {
  union { unsigned u[8]; v16bf v; } f;
  const unsigned* p32 = (const unsigned*)base;
  int row = lane & 15;
  int kh  = (lane >> 4) << 3;          // K-base in uint units (0 or 8)
  int rb  = row * (stride >> 1);
  #pragma unroll
  for (int i = 0; i < 4; ++i) f.u[i]     = p32[rb + kh + i];
  #pragma unroll
  for (int i = 0; i < 4; ++i) f.u[4 + i] = p32[rb + kh + 4 + i];
  return f.v;
}

// B-matrix fragment (32x16 bf16) from K-contiguous storage: column n at base+n*stride.
// lane<16 -> N=lane, K=0..15 ; lane>=16 -> N=lane-16, K=16..31
__device__ inline v16bf load_b_frag(const bf16_t* base, int stride, int lane) {
  union { unsigned u[8]; v16bf v; } f;
  const unsigned* p32 = (const unsigned*)base;
  int off = (((lane & 15) * stride) + ((lane >> 4) << 4)) >> 1;
  #pragma unroll
  for (int i = 0; i < 8; ++i) f.u[i] = p32[off + i];
  return f.v;
}

// ---------------------------------------------------------------------------
// fp32 -> bf16 cast (row-major copy)
// ---------------------------------------------------------------------------
__global__ void cast_f32_bf16(const float* __restrict__ in, bf16_t* __restrict__ out, int n) {
  int i = blockIdx.x * 256 + threadIdx.x;
  if (i < n) out[i] = (bf16_t)in[i];
}

// fp32 [R][C] -> bf16 transposed [C][R] (one block per output row; coalesced stores)
__global__ void transpose_cast_kernel(const float* __restrict__ in, bf16_t* __restrict__ out,
                                      int R, int Ccols) {
  int n = blockIdx.x;                          // output row = input column
  bf16_t* dst = out + (size_t)n * R;
  for (int k = threadIdx.x; k < R; k += 256)
    dst[k] = (bf16_t)in[(size_t)k * Ccols + n];
}

// V transpose: qkv[n][2*D + h*80 + d] -> vT[(h*80+d)][n]  (token-contiguous columns)
__global__ void v_transpose_kernel(const bf16_t* __restrict__ qkv, bf16_t* __restrict__ vT) {
  int hd = blockIdx.x;                         // h*HD + d, 0..1279
  const bf16_t* src = qkv + 2 * DMODEL + hd;
  bf16_t* dst = vT + (size_t)hd * NTOK;
  for (int n = threadIdx.x; n < NTOK; n += 256)
    dst[n] = src[(size_t)n * 3 * DMODEL];
}

// ---------------------------------------------------------------------------
// bf16 GEMM: C[M,N] = A[M,K] * Bt[N,K]^T + bias[N].  128x128 block, 8 waves.
// A double-buffered in LDS via async DMA; B fragments straight from global.
// ---------------------------------------------------------------------------
template <typename OutT, int KD>
__global__ __launch_bounds__(256) void gemm_bf16_kernel(
    const bf16_t* __restrict__ A,      // [M][KD] row-major
    const bf16_t* __restrict__ Bt,     // [N][KD] row-major (pre-transposed weights)
    const float* __restrict__ bias, OutT* __restrict__ C,
    int Mdim, int Ndim)
{
  constexpr int ST = 40;               // padded LDS row stride (elements)
  __shared__ bf16_t sA[2][128 * ST];
  const int tid = threadIdx.x, wave = tid >> 5, lane = tid & 31;
  const int mB = blockIdx.y * 128, nB = blockIdx.x * 128;
  const int wRow = (wave & 1) * 64, wCol = (wave >> 1) * 32;

  auto stage = [&](int kk, int buf) {
    #pragma unroll
    for (int t = 0; t < 2; ++t) {
      int s = tid + t * 256;               // 512 x b128 covers 128x32 tile
      int r = s >> 2, c = (s & 3) << 3;
      unsigned lds_off = (unsigned)(size_t)&sA[buf][r * ST + c];
      async_load_b128(lds_off, A + (size_t)(mB + r) * KD + kk + c);
    }
  };

  v8f acc[4][2];
  #pragma unroll
  for (int mt = 0; mt < 4; ++mt)
    #pragma unroll
    for (int nt = 0; nt < 2; ++nt) acc[mt][nt] = vzero8();

  stage(0, 0);
  wait_asynccnt0();
  __syncthreads();

  int buf = 0;
  for (int kk = 0; kk < KD; kk += 32) {
    if (kk + 32 < KD) stage(kk + 32, buf ^ 1);

    v16bf aF[4], bF[2];
    #pragma unroll
    for (int mt = 0; mt < 4; ++mt)
      aF[mt] = load_a_frag(sA[buf] + (wRow + mt * 16) * ST, ST, lane);
    #pragma unroll
    for (int nt = 0; nt < 2; ++nt)
      bF[nt] = load_b_frag(Bt + (size_t)(nB + wCol + nt * 16) * KD + kk, KD, lane);
    #pragma unroll
    for (int mt = 0; mt < 4; ++mt)
      #pragma unroll
      for (int nt = 0; nt < 2; ++nt)
        acc[mt][nt] = WMMA_BF16(aF[mt], bF[nt], acc[mt][nt]);

    wait_asynccnt0();
    __syncthreads();
    buf ^= 1;
  }

  // Epilogue: C layout lane -> (M = v + (lane>>4)*8, N = lane&15)
  #pragma unroll
  for (int mt = 0; mt < 4; ++mt) {
    #pragma unroll
    for (int nt = 0; nt < 2; ++nt) {
      int gn = nB + wCol + nt * 16 + (lane & 15);
      float bv = bias[gn];
      #pragma unroll
      for (int v = 0; v < 8; ++v) {
        int gm = mB + wRow + mt * 16 + ((lane >> 4) << 3) + v;
        C[(size_t)gm * Ndim + gn] = (OutT)(acc[mt][nt][v] + bv);
      }
    }
  }
}

// ---------------------------------------------------------------------------
// RoPE + head layout: qkv[N][3*D] -> qp/kp[N][NH][HDP] (bf16, zero-padded 80->96)
// ---------------------------------------------------------------------------
__global__ void rope_kernel(const bf16_t* __restrict__ qkv,
                            const float* __restrict__ cosT, const float* __restrict__ sinT,
                            bf16_t* __restrict__ qp, bf16_t* __restrict__ kp)
{
  int b = blockIdx.x;          // n*NH + h
  int n = b >> 4, h = b & 15;
  int dp = threadIdx.x;        // 0..127
  if (dp >= HDP) return;
  size_t ob = (size_t)b * HDP + dp;
  if (dp < HD) {
    const bf16_t* row = qkv + (size_t)n * (3 * DMODEL) + h * HD;
    float qv = (float)row[dp];
    float kv = (float)row[DMODEL + dp];
    int d2 = (dp < 40) ? dp + 40 : dp - 40;
    float qr = (dp < 40) ? -(float)row[d2] : (float)row[d2];
    float kr = (dp < 40) ? -(float)row[DMODEL + d2] : (float)row[DMODEL + d2];
    float c = cosT[n * HD + dp], s = sinT[n * HD + dp];
    qp[ob] = (bf16_t)(qv * c + qr * s);
    kp[ob] = (bf16_t)(kv * c + kr * s);
  } else {
    qp[ob] = (bf16_t)0.f;
    kp[ob] = (bf16_t)0.f;
  }
}

// ---------------------------------------------------------------------------
// Varlen attention: one workgroup per (16-query tile, head)
// ---------------------------------------------------------------------------
__global__ __launch_bounds__(256) void attn_kernel(
    const bf16_t* __restrict__ qp, const bf16_t* __restrict__ kp,
    const bf16_t* __restrict__ vT, const int* __restrict__ cu,
    bf16_t* __restrict__ attn)
{
  __shared__ float  S[16 * SPAD];      // fp32 scores
  __shared__ bf16_t Pb[16 * PPAD];     // bf16 softmax probabilities
  __shared__ float  red[256];
  __shared__ float  rmax[16];
  __shared__ float  rinv[16];

  const int tid = threadIdx.x, wave = tid >> 5, lane = tid & 31;
  const int h = blockIdx.y;

  int b = blockIdx.x, m, L = 0, base = 0;
  for (m = 0; m < NSEQ; ++m) {
    base = cu[m];
    L = cu[m + 1] - base;
    int nt = L >> 4;
    if (b < nt) break;
    b -= nt;
  }
  if (m >= NSEQ) return;
  const int q0 = base + (b << 4);

  // Q fragments: 3 chunks of 32 over padded head dim 96 (registers, all waves)
  v16bf qF[3];
  {
    const bf16_t* qb = qp + ((size_t)q0 * NH + h) * HDP;
    #pragma unroll
    for (int c = 0; c < 3; ++c) qF[c] = load_a_frag(qb + c * 32, NH * HDP, lane);
  }

  // Scores S = scale * Q K^T  (wave w covers key chunks w*32, w*32+256, ...)
  for (int kc = wave * 32; kc < L; kc += 256) {
    v8f s0 = vzero8(), s1 = vzero8();
    const bf16_t* kb0 = kp + ((size_t)(base + kc) * NH + h) * HDP;
    const bf16_t* kb1 = kp + ((size_t)(base + kc + 16) * NH + h) * HDP;
    #pragma unroll
    for (int c = 0; c < 3; ++c) {
      v16bf bf0 = load_b_frag(kb0 + c * 32, NH * HDP, lane);
      v16bf bf1 = load_b_frag(kb1 + c * 32, NH * HDP, lane);
      s0 = WMMA_BF16(qF[c], bf0, s0);
      s1 = WMMA_BF16(qF[c], bf1, s1);
    }
    int row = (lane >> 4) << 3;
    int col = lane & 15;
    #pragma unroll
    for (int v = 0; v < 8; ++v) {
      S[(row + v) * SPAD + kc + col]      = s0[v] * QSCALE;
      S[(row + v) * SPAD + kc + 16 + col] = s1[v] * QSCALE;
    }
  }
  __syncthreads();

  // Two-pass softmax per query row (16 threads/row); P written as bf16
  {
    int row = tid >> 4, sub = tid & 15;
    float mx = -3.0e38f;
    for (int j = sub; j < L; j += 16) mx = fmaxf(mx, S[row * SPAD + j]);
    red[row * 16 + sub] = mx;
    __syncthreads();
    if (sub == 0) {
      float v = red[row * 16];
      for (int i = 1; i < 16; ++i) v = fmaxf(v, red[row * 16 + i]);
      rmax[row] = v;
    }
    __syncthreads();
    float rm = rmax[row];
    float sum = 0.f;
    for (int j = sub; j < L; j += 16) {
      float p = __expf(S[row * SPAD + j] - rm);
      sum += p;
      Pb[row * PPAD + j] = (bf16_t)p;
    }
    red[row * 16 + sub] = sum;
    __syncthreads();
    if (sub == 0) {
      float v = 0.f;
      for (int i = 0; i < 16; ++i) v += red[row * 16 + i];
      rinv[row] = 1.f / v;
    }
    __syncthreads();
  }

  // O = P V (5 head-dim tiles of 16 on waves 0..4; 1/sum folded into epilogue)
  if (wave < 5) {
    int dt = wave * 16;
    v8f o = vzero8();
    const bf16_t* vb = vT + (size_t)(h * HD + dt) * NTOK + base;
    for (int kc = 0; kc < L; kc += 32) {
      v16bf pF = load_a_frag(Pb + kc, PPAD, lane);
      v16bf vF = load_b_frag(vb + kc, NTOK, lane);
      o = WMMA_BF16(pF, vF, o);
    }
    int row = (lane >> 4) << 3;
    int col = lane & 15;
    #pragma unroll
    for (int v = 0; v < 8; ++v) {
      int qrow = row + v;
      float val = o[v] * rinv[qrow];
      attn[(size_t)(q0 + qrow) * DMODEL + h * HD + dt + col] = (bf16_t)val;
    }
  }
}

// ---------------------------------------------------------------------------
extern "C" void kernel_launch(void* const* d_in, const int* in_sizes, int n_in,
                              void* d_out, int out_size, void* d_ws, size_t ws_size,
                              hipStream_t stream)
{
  (void)in_sizes; (void)n_in; (void)out_size; (void)ws_size;
  const float* hidden = (const float*)d_in[0];
  const int*   cu     = (const int*)d_in[1];
  const float* cosT   = (const float*)d_in[2];
  const float* sinT   = (const float*)d_in[3];
  const float* Wqkv   = (const float*)d_in[4];
  const float* bqkv   = (const float*)d_in[5];
  const float* Wproj  = (const float*)d_in[6];
  const float* bproj  = (const float*)d_in[7];
  float* out = (float*)d_out;

  char* ws = (char*)d_ws;
  size_t o = 0;
  bf16_t* hbf   = (bf16_t*)(ws + o); o += (size_t)NTOK * DMODEL * 2;
  bf16_t* wqT   = (bf16_t*)(ws + o); o += (size_t)3 * DMODEL * DMODEL * 2;   // [3D][D]
  bf16_t* wpT   = (bf16_t*)(ws + o); o += (size_t)DMODEL * DMODEL * 2;       // [D][D]
  bf16_t* qkvbf = (bf16_t*)(ws + o); o += (size_t)NTOK * 3 * DMODEL * 2;
  bf16_t* qpb   = (bf16_t*)(ws + o); o += (size_t)NTOK * NH * HDP * 2;
  bf16_t* kpb   = (bf16_t*)(ws + o); o += (size_t)NTOK * NH * HDP * 2;
  bf16_t* vTb   = (bf16_t*)(ws + o); o += (size_t)NH * HD * NTOK * 2;        // [h*80+d][n]
  bf16_t* attnb = (bf16_t*)(ws + o); o += (size_t)NTOK * DMODEL * 2;

  int n1 = NTOK * DMODEL;
  cast_f32_bf16<<<(n1 + 255) / 256, 256, 0, stream>>>(hidden, hbf, n1);
  transpose_cast_kernel<<<3 * DMODEL, 256, 0, stream>>>(Wqkv, wqT, DMODEL, 3 * DMODEL);
  transpose_cast_kernel<<<DMODEL, 256, 0, stream>>>(Wproj, wpT, DMODEL, DMODEL);

  // qkv = hidden @ W_qkv + b_qkv   (8192 x 3840, K=1280)
  gemm_bf16_kernel<bf16_t, DMODEL><<<dim3(3 * DMODEL / 128, NTOK / 128), 256, 0, stream>>>(
      hbf, wqT, bqkv, qkvbf, NTOK, 3 * DMODEL);

  // RoPE + head layout; V transposed to token-contiguous columns
  rope_kernel<<<NTOK * NH, 128, 0, stream>>>(qkvbf, cosT, sinT, qpb, kpb);
  v_transpose_kernel<<<NH * HD, 256, 0, stream>>>(qkvbf, vTb);

  // attention: 512 query tiles (N/16) x 16 heads
  attn_kernel<<<dim3(NTOK / 16, NH), 256, 0, stream>>>(qpb, kpb, vTb, cu, attnb);

  // out = attn @ W_proj + b_proj   (8192 x 1280, K=1280)
  gemm_bf16_kernel<float, DMODEL><<<dim3(DMODEL / 128, NTOK / 128), 256, 0, stream>>>(
      attnb, wpT, bproj, out, NTOK, DMODEL);
}
